// SurfaceVAE_5196910428284
// MI455X (gfx1250) — matrix operand order repaired and checked by reading, not compiled, not run on hardware
//
#include <hip/hip_runtime.h>
#include <hip/hip_bf16.h>

typedef __attribute__((ext_vector_type(16))) _Float16 v16h;
typedef __attribute__((ext_vector_type(8)))  float    v8f;

// ---------------------------------------------------------------------------
// Packed-weight layout (built by vae_prep): for a weight [K][N] (K multiple of
// 32), fragment tile t2 = nt*KC + kc holds the WMMA B-fragment for k-chunk kc,
// n-tile nt, stored lane-major: dst[(t2*32 + lane)*16 + e] where
//   k = kc*32 + (lane>>4)*16 + e,  n = nt*16 + (lane&15)
// so each lane's 16 halfs are one contiguous 32-byte v16h load.
// ---------------------------------------------------------------------------

__device__ __forceinline__ void pack_w(_Float16* dst, const float* src, int idx,
                                       int KC, int Kreal, int N) {
    int e    = idx & 15;
    int lane = (idx >> 4) & 31;
    int t2   = idx >> 9;
    int nt   = t2 / KC;
    int kc   = t2 - nt * KC;
    int k    = kc * 32 + (lane >> 4) * 16 + e;
    int n    = nt * 16 + (lane & 15);
    dst[idx] = (k < Kreal) ? (_Float16)src[k * N + n] : (_Float16)0.0f;
}

// region offsets in halfs within d_ws
#define OFF_W1P   0         // [64][512]  KC=2  (K real 48)
#define OFF_W2P   32768     // [512][256] KC=16
#define OFF_W3P   163840    // [256][128] KC=8
#define OFF_WMLV  196608    // [128][256] KC=4  (mu|lv packed)
#define OFF_WD1P  229376    // [160][256] KC=5  (K real 144)
#define OFF_WD2P  270336    // [256][512] KC=8
#define OFF_WD3P  401408    // [512][32]  KC=16
#define WS_HALFS  417792
#define PREP_TOTAL (WS_HALFS + 256)

__global__ void vae_prep(const float* W1, const float* W2, const float* W3,
                         const float* Wmu, const float* Wlv,
                         const float* bmu, const float* blv,
                         const float* Wd1, const float* Wd2, const float* Wd3,
                         _Float16* ws16, float* bias_mlv, int total) {
    for (int idx = blockIdx.x * blockDim.x + threadIdx.x; idx < total;
         idx += gridDim.x * blockDim.x) {
        if (idx < OFF_W2P) {
            pack_w(ws16 + OFF_W1P, W1, idx - OFF_W1P, 2, 48, 512);
        } else if (idx < OFF_W3P) {
            pack_w(ws16 + OFF_W2P, W2, idx - OFF_W2P, 16, 512, 256);
        } else if (idx < OFF_WMLV) {
            pack_w(ws16 + OFF_W3P, W3, idx - OFF_W3P, 8, 256, 128);
        } else if (idx < OFF_WD1P) {
            int li = idx - OFF_WMLV;
            int e = li & 15, lane = (li >> 4) & 31, t2 = li >> 9;
            int nt = t2 / 4, kc = t2 - nt * 4;
            int k = kc * 32 + (lane >> 4) * 16 + e;
            int n = nt * 16 + (lane & 15);
            float v = (n < 128) ? Wmu[k * 128 + n] : Wlv[k * 128 + (n - 128)];
            ws16[idx] = (_Float16)v;
        } else if (idx < OFF_WD2P) {
            pack_w(ws16 + OFF_WD1P, Wd1, idx - OFF_WD1P, 5, 144, 256);
        } else if (idx < OFF_WD3P) {
            pack_w(ws16 + OFF_WD2P, Wd2, idx - OFF_WD2P, 8, 256, 512);
        } else if (idx < WS_HALFS) {
            pack_w(ws16 + OFF_WD3P, Wd3, idx - OFF_WD3P, 16, 512, 32);
        } else {
            int c = idx - WS_HALFS;  // 0..255  mu bias | lv bias
            bias_mlv[c] = (c < 128) ? bmu[c] : blv[c - 128];
        }
    }
}

// ---------------------------------------------------------------------------
// Per-block fused GEMM: M=64 (4 m-tiles), 8 waves. wave&3 -> m-tile,
// wave>>2 -> n-tile parity. Activations in LDS, stride 512 halfs.
// ---------------------------------------------------------------------------
template <int K, int N, bool RELU, bool MULV>
__device__ __forceinline__ void gemm_tile(const _Float16* A, const _Float16* Wp,
                                          const float* __restrict__ bias,
                                          _Float16* D, int wave, int lane,
                                          float* gmu, float* glv, int rowbase) {
    constexpr int KC = K / 32;
    constexpr int NT = N / 16;
    constexpr int JT = NT / 2;
    const int mtile = wave & 3;
    const int npar  = wave >> 2;
    const int l16   = lane & 15;
    const int hsel  = lane >> 4;
    const int arow  = mtile * 16 + l16;

    v8f acc[JT];
    v8f zeroacc = {};
#pragma unroll
    for (int j = 0; j < JT; ++j) acc[j] = zeroacc;

    for (int kc = 0; kc < KC; ++kc) {
        // A fragment (16x32 f16): lanes 0-15 row M, K 0-15; lanes 16-31 K 16-31
        v16h a;
#pragma unroll
        for (int i = 0; i < 8; ++i) {
            int kk = kc * 32 + hsel * 16 + i * 2;
            a[2 * i]     = A[arow * 512 + kk];
            a[2 * i + 1] = A[arow * 512 + kk + 1];
        }
#pragma unroll
        for (int j = 0; j < JT; ++j) {
            int nt = 2 * j + npar;
            v16h b = *(const v16h*)(Wp + (size_t)((nt * KC + kc) * 32 + lane) * 16);
            acc[j] = __builtin_amdgcn_wmma_f32_16x16x32_f16(
                false, a, false, b, (short)0, acc[j], false, false);
        }
    }
    // epilogue: C/D layout -> D[m][n] in LDS (+ optional exact f32 mu/lv out)
#pragma unroll
    for (int j = 0; j < JT; ++j) {
        int n0 = (2 * j + npar) * 16;
        float bv = bias[n0 + l16];
#pragma unroll
        for (int i = 0; i < 8; ++i) {
            int m = mtile * 16 + hsel * 8 + i;
            float v = acc[j][i] + bv;
            if (RELU) v = fmaxf(v, 0.0f);
            D[m * 512 + n0 + l16] = (_Float16)v;
            if (MULV) {
                int n = n0 + l16;
                if (n < 128) gmu[(size_t)(rowbase + m) * 128 + n] = v;
                else         glv[(size_t)(rowbase + m) * 128 + (n - 128)] = v;
            }
        }
    }
}

__global__ __launch_bounds__(256) void vae_main(
    const float* __restrict__ params_raw, const int* __restrict__ stype_g,
    const float* __restrict__ eps, const float* __restrict__ type_emb,
    const float* __restrict__ W_pe, const float* __restrict__ b_pe,
    const float* __restrict__ b1, const float* __restrict__ b2,
    const float* __restrict__ b3, const float* __restrict__ bd1,
    const float* __restrict__ bd2, const float* __restrict__ bd3,
    const float* __restrict__ Wc, const float* __restrict__ bc,
    const float* __restrict__ W_dr, const float* __restrict__ b_dr,
    const _Float16* __restrict__ w1p, const _Float16* __restrict__ w2p,
    const _Float16* __restrict__ w3p, const _Float16* __restrict__ wmlv,
    const float* __restrict__ bias_mlv, const _Float16* __restrict__ wd1p,
    const _Float16* __restrict__ wd2p, const _Float16* __restrict__ wd3p,
    float* __restrict__ out_recon, float* __restrict__ out_mask,
    float* __restrict__ out_logits, float* __restrict__ out_mu,
    float* __restrict__ out_lv) {
    extern __shared__ char smem[];
    _Float16* bufA = (_Float16*)smem;              // 64 x 512 halfs
    _Float16* bufB = bufA + 64 * 512;              // 64 x 512 halfs
    int* s_type = (int*)(bufB + 64 * 512);         // 64 ints

    const int rowbase = blockIdx.x * 64;
    const int wave = threadIdx.x >> 5;
    const int lane = threadIdx.x & 31;

    // ---- Phase A: per-type input embedding (VALU): x = [param_emb | emb | 0]
    {
        int r = threadIdx.x >> 2, q = threadIdx.x & 3;
        int grow = rowbase + r;
        int t = stype_g[grow];
        if (q == 0) s_type[r] = t;
        float raw[17];
#pragma unroll
        for (int d = 0; d < 17; ++d) raw[d] = params_raw[(size_t)grow * 17 + d];
#pragma unroll
        for (int pp = 0; pp < 8; ++pp) {
            int p = q * 8 + pp;
            float s = b_pe[t * 32 + p];
#pragma unroll
            for (int d = 0; d < 17; ++d) s += raw[d] * W_pe[(t * 17 + d) * 32 + p];
            bufA[r * 512 + p] = (_Float16)s;
        }
#pragma unroll
        for (int i = 0; i < 4; ++i) {
            bufA[r * 512 + 32 + q * 4 + i] = (_Float16)type_emb[t * 16 + q * 4 + i];
            bufA[r * 512 + 48 + q * 4 + i] = (_Float16)0.0f;
        }
    }
    __syncthreads();

    // ---- encoder MLP (WMMA) ----
    gemm_tile<64, 512, true, false>(bufA, w1p, b1, bufB, wave, lane, nullptr, nullptr, rowbase);
    __syncthreads();
    gemm_tile<512, 256, true, false>(bufB, w2p, b2, bufA, wave, lane, nullptr, nullptr, rowbase);
    __syncthreads();
    gemm_tile<256, 128, false, false>(bufA, w3p, b3, bufB, wave, lane, nullptr, nullptr, rowbase);
    __syncthreads();
    // mu|logvar fused GEMM, exact f32 written to global in epilogue
    gemm_tile<128, 256, false, true>(bufB, wmlv, bias_mlv, bufA, wave, lane, out_mu, out_lv, rowbase);
    __syncthreads();

    // ---- Phase F: reparameterize, build xd = [z | emb | 0]
    {
        int r = threadIdx.x >> 2, q = threadIdx.x & 3;
        int grow = rowbase + r;
        int t = s_type[r];
#pragma unroll
        for (int i = 0; i < 32; ++i) {
            int c = q * 32 + i;
            float m = (float)bufA[r * 512 + c];
            float l = (float)bufA[r * 512 + 128 + c];
            float lc = fminf(fmaxf(l, -10.0f), 10.0f);
            float zv = m + eps[(size_t)grow * 128 + c] * __expf(0.5f * lc);
            bufB[r * 512 + c] = (_Float16)zv;
        }
#pragma unroll
        for (int i = 0; i < 4; ++i) {
            bufB[r * 512 + 128 + q * 4 + i] = (_Float16)type_emb[t * 16 + q * 4 + i];
            bufB[r * 512 + 144 + q * 4 + i] = (_Float16)0.0f;
        }
    }
    __syncthreads();

    // ---- Phase G: classifier z @ Wc + bc (tiny, VALU)
    if (threadIdx.x < 64) {
        int r = threadIdx.x;
        int grow = rowbase + r;
        float acc[5];
#pragma unroll
        for (int t = 0; t < 5; ++t) acc[t] = bc[t];
        for (int c = 0; c < 128; ++c) {
            float zz = (float)bufB[r * 512 + c];
#pragma unroll
            for (int t = 0; t < 5; ++t) acc[t] += zz * Wc[c * 5 + t];
        }
#pragma unroll
        for (int t = 0; t < 5; ++t) out_logits[(size_t)grow * 5 + t] = acc[t];
    }

    // ---- decoder MLP (WMMA) ----
    gemm_tile<160, 256, true, false>(bufB, wd1p, bd1, bufA, wave, lane, nullptr, nullptr, rowbase);
    __syncthreads();
    gemm_tile<256, 512, true, false>(bufA, wd2p, bd2, bufB, wave, lane, nullptr, nullptr, rowbase);
    __syncthreads();
    gemm_tile<512, 32, false, false>(bufB, wd3p, bd3, bufA, wave, lane, nullptr, nullptr, rowbase);
    __syncthreads();

    // ---- Phase K: per-type raw decode + mask (VALU)
    {
        int r = threadIdx.x >> 2, q = threadIdx.x & 3;
        int grow = rowbase + r;
        int t = s_type[r];
        const int dims_arr[5] = {7, 9, 12, 15, 17};
        int dims = dims_arr[t];
        float pe[32];
#pragma unroll
        for (int p = 0; p < 32; ++p) pe[p] = (float)bufA[r * 512 + p];
        int jend = q * 5 + 5; if (jend > 17) jend = 17;
        for (int j = q * 5; j < jend; ++j) {
            float s = b_dr[t * 17 + j];
#pragma unroll
            for (int p = 0; p < 32; ++p) s += pe[p] * W_dr[(t * 32 + p) * 17 + j];
            out_recon[(size_t)grow * 17 + j] = s;
            out_mask[(size_t)grow * 17 + j] = (j < dims) ? 1.0f : 0.0f;
        }
    }
}

extern "C" void kernel_launch(void* const* d_in, const int* in_sizes, int n_in,
                              void* d_out, int out_size, void* d_ws, size_t ws_size,
                              hipStream_t stream) {
    (void)n_in; (void)out_size; (void)ws_size;
    const float* params_raw = (const float*)d_in[0];
    const int*   stype      = (const int*)d_in[1];
    const float* eps        = (const float*)d_in[2];
    const float* type_emb   = (const float*)d_in[3];
    const float* W_pe = (const float*)d_in[4];  const float* b_pe = (const float*)d_in[5];
    const float* W1   = (const float*)d_in[6];  const float* b1   = (const float*)d_in[7];
    const float* W2   = (const float*)d_in[8];  const float* b2   = (const float*)d_in[9];
    const float* W3   = (const float*)d_in[10]; const float* b3   = (const float*)d_in[11];
    const float* Wmu  = (const float*)d_in[12]; const float* bmu  = (const float*)d_in[13];
    const float* Wlv  = (const float*)d_in[14]; const float* blv  = (const float*)d_in[15];
    const float* Wd1  = (const float*)d_in[16]; const float* bd1  = (const float*)d_in[17];
    const float* Wd2  = (const float*)d_in[18]; const float* bd2  = (const float*)d_in[19];
    const float* Wd3  = (const float*)d_in[20]; const float* bd3  = (const float*)d_in[21];
    const float* Wc   = (const float*)d_in[22]; const float* bc   = (const float*)d_in[23];
    const float* W_dr = (const float*)d_in[24]; const float* b_dr = (const float*)d_in[25];

    const int B = in_sizes[1];  // surface_type count == batch

    _Float16* ws16 = (_Float16*)d_ws;
    float* bias_mlv = (float*)((char*)d_ws + (size_t)WS_HALFS * sizeof(_Float16));

    vae_prep<<<512, 256, 0, stream>>>(W1, W2, W3, Wmu, Wlv, bmu, blv, Wd1, Wd2, Wd3,
                                      ws16, bias_mlv, PREP_TOTAL);

    float* out = (float*)d_out;
    float* out_recon  = out;
    float* out_mask   = out + (size_t)B * 17;
    float* out_logits = out + (size_t)B * 34;
    float* out_mu     = out + (size_t)B * 39;
    float* out_lv     = out + (size_t)B * 167;

    size_t shmem = (size_t)64 * 512 * 2 * sizeof(_Float16) + 64 * sizeof(int);
    vae_main<<<B / 64, 256, shmem, stream>>>(
        params_raw, stype, eps, type_emb, W_pe, b_pe,
        b1, b2, b3, bd1, bd2, bd3, Wc, bc, W_dr, b_dr,
        ws16 + OFF_W1P, ws16 + OFF_W2P, ws16 + OFF_W3P, ws16 + OFF_WMLV, bias_mlv,
        ws16 + OFF_WD1P, ws16 + OFF_WD2P, ws16 + OFF_WD3P,
        out_recon, out_mask, out_logits, out_mu, out_lv);
}
